// CustomTransformerEncoderMoELayer_20418274525445
// MI455X (gfx1250) — matrix operand order repaired and checked by reading, not compiled, not run on hardware
//
#include <hip/hip_runtime.h>

// ---------------------------------------------------------------------------
// CDNA5 (gfx1250) wave32 WMMA implementation of the MoE transformer layer.
// All matrix math runs through v_wmma_f32_16x16x32_bf16 (bf16 in, f32 acc).
// ---------------------------------------------------------------------------

typedef __attribute__((ext_vector_type(16))) __bf16 v16bf;
typedef __attribute__((ext_vector_type(8)))  float  v8f;
typedef __attribute__((ext_vector_type(4)))  int    i4;
typedef unsigned short u16;

union BfPack { i4 q[2]; v16bf v; };
static_assert(sizeof(BfPack) == 32, "bfpack");

static __device__ inline v8f wmma_bf16(v16bf a, v16bf b, v8f c) {
  // (neg_a, A, neg_b, B, c_mod, C, reuse_a, reuse_b)
  return __builtin_amdgcn_wmma_f32_16x16x32_bf16(false, a, false, b, (short)0, c,
                                                 false, false);
}

static __device__ inline u16 f2bf(float f) {   // round-to-nearest-even
  unsigned int u = __float_as_uint(f);
  u += 0x7FFFu + ((u >> 16) & 1u);
  return (u16)(u >> 16);
}

// ---------------------------------------------------------------------------
// elementwise helpers
// ---------------------------------------------------------------------------
__global__ __launch_bounds__(256) void cvt_bf16_kernel(const float* __restrict__ s,
                                                       u16* __restrict__ d, size_t n) {
  size_t i = (size_t)blockIdx.x * 256 + threadIdx.x;
  if (i < n) d[i] = f2bf(s[i]);
}

__global__ __launch_bounds__(256) void zero_f32_kernel(float* __restrict__ p, size_t n) {
  size_t i = (size_t)blockIdx.x * 256 + threadIdx.x;
  if (i < n) p[i] = 0.f;
}

// src [rows][cols] f32 -> dst [cols][rows] bf16  (blockIdx.z = batch)
__global__ __launch_bounds__(256) void transpose_bf16_kernel(const float* __restrict__ src,
                                                             u16* __restrict__ dst,
                                                             int rows, int cols) {
  size_t n = (size_t)rows * cols;
  src += (size_t)blockIdx.z * n;
  dst += (size_t)blockIdx.z * n;
  size_t i = (size_t)blockIdx.x * 256 + threadIdx.x;
  if (i >= n) return;
  size_t k = i % rows, c = i / rows;
  dst[c * (size_t)rows + k] = f2bf(src[k * (size_t)cols + c]);
}

// ---------------------------------------------------------------------------
// Tiled bf16 GEMM: C[M,N] = A[M,K] x Bt[N,K]^T (+bias)  via WMMA 16x16x32.
// Block = 256 threads = 8 waves; tile 128(M) x 128(N), K-step 32.
// Waves arranged 4(M) x 2(N): each wave owns 32x64 = 2x4 WMMA C tiles
// -> 8 v_wmma per wave per K-step between barriers.
// Global->LDS staging is register double-buffered: next K-step's fragments
// are fetched right after the first barrier so the b128 loads overlap the
// WMMAs. LDS row pitch 80B: 16B-aligned, bank-conflict-free ds_load_b128
// operand fetches matching the ISA 7.12.2 A/B VGPR layouts.
// MODE: 0 f32 out (+bias); 1 bf16 head layout [B,H,T,64] (Q,K);
//       2 bf16 V^T layout [B,H,64,T]; 3 relu->bf16, gathered A (MoE W1);
//       4 weighted atomic scatter into f32 rows (MoE W2 combine).
// ---------------------------------------------------------------------------
template <int MODE, bool GATHER>
__global__ __launch_bounds__(256) void gemm_bf16_kernel(
    const u16* __restrict__ A, const u16* __restrict__ Bt,
    const float* __restrict__ bias, float* __restrict__ outf,
    u16* __restrict__ outbf, int N, int K,
    const int* __restrict__ expert_off, const int* __restrict__ rowlist,
    const float* __restrict__ wlist) {
  __shared__ __attribute__((aligned(16))) unsigned char ldsA[128 * 80];
  __shared__ __attribute__((aligned(16))) unsigned char ldsB[128 * 80];
  __shared__ int   rlLds[128];
  __shared__ float wlLds[128];

  const int tid = threadIdx.x, wid = tid >> 5, lane = tid & 31;
  const int sel = lane >> 4, ln = lane & 15;
  const int wm = wid & 3, wn = wid >> 2;
  const int nbase = blockIdx.x * 128;

  int mbase, valid = 128, ex = 0;
  if constexpr (MODE >= 3) {
    ex = blockIdx.z;
    int start = expert_off[ex], end = expert_off[ex + 1];
    mbase = start + blockIdx.y * 128;
    if (mbase >= end) return;                 // block-uniform exit (EXEC stays full)
    valid = end - mbase; if (valid > 128) valid = 128;
  } else {
    mbase = blockIdx.y * 128;
  }
  const u16*   Btb   = Bt + (size_t)ex * N * K;
  const float* biasp = bias + (size_t)ex * N;

  if constexpr (GATHER || MODE == 4) {
    if (tid < 128) {
      int ok = tid < valid;
      rlLds[tid] = ok ? rowlist[mbase + tid] : 0;
      if constexpr (MODE == 4) wlLds[tid] = ok ? wlist[mbase + tid] : 0.f;
    }
  }
  __syncthreads();

  // 128 rows x 32 k of bf16 = 512 x 16B chunks; 2 chunks per thread per tile.
  const int r0 = tid >> 2, r1 = (tid + 256) >> 2;   // chunk rows
  const int g0 = (tid & 3) * 8, g1 = g0;            // k-seg (elements)
  const int arowA0 = GATHER ? 0 : mbase + r0;       // placeholders, resolved below

  i4 stA[2], stB[2];
  auto stage_load = [&](int k0) {
    int ra0 = GATHER ? rlLds[r0] : (mbase + r0);
    int ra1 = GATHER ? rlLds[r1] : (mbase + r1);
    stA[0] = *(const i4*)(A + (size_t)ra0 * K + k0 + g0);
    stA[1] = *(const i4*)(A + (size_t)ra1 * K + k0 + g1);
    const u16* bp0 = Btb + (size_t)(nbase + r0) * K + k0 + g0;
    const u16* bp1 = Btb + (size_t)(nbase + r1) * K + k0 + g1;
    stB[0] = *(const i4*)bp0;
    stB[1] = *(const i4*)bp1;
    if (k0 + 64 < K) {                               // gfx1250 global_prefetch
      __builtin_prefetch(bp0 + 32, 0, 3);
      __builtin_prefetch(bp1 + 32, 0, 3);
    }
  };
  (void)arowA0;

  stage_load(0);

  v8f c[2][4] = {};
  for (int k0 = 0; k0 < K; k0 += 32) {
    // commit staged fragments to LDS
    *(i4*)(ldsA + r0 * 80 + (g0 << 1)) = stA[0];
    *(i4*)(ldsA + r1 * 80 + (g1 << 1)) = stA[1];
    *(i4*)(ldsB + r0 * 80 + (g0 << 1)) = stB[0];
    *(i4*)(ldsB + r1 * 80 + (g1 << 1)) = stB[1];
    __syncthreads();

    if (k0 + 32 < K) stage_load(k0 + 32);   // overlaps with the WMMAs below

    BfPack a[2];
#pragma unroll
    for (int i = 0; i < 2; ++i) {            // A-operand layout
      const unsigned char* ap = ldsA + (wm * 32 + i * 16 + ln) * 80 + sel * 16;
      a[i].q[0] = *(const i4*)ap;
      a[i].q[1] = *(const i4*)(ap + 32);
    }
#pragma unroll
    for (int j = 0; j < 4; ++j) {            // B-operand layout
      BfPack bo;
      const unsigned char* bp = ldsB + (wn * 64 + j * 16 + ln) * 80 + sel * 16;
      bo.q[0] = *(const i4*)bp;
      bo.q[1] = *(const i4*)(bp + 32);
      c[0][j] = wmma_bf16(a[0].v, bo.v, c[0][j]);
      c[1][j] = wmma_bf16(a[1].v, bo.v, c[1][j]);
    }
    __syncthreads();
  }

#pragma unroll
  for (int i = 0; i < 2; ++i)
#pragma unroll
    for (int j = 0; j < 4; ++j) {
      int col = nbase + wn * 64 + j * 16 + ln;
      float bvl = biasp[col];
#pragma unroll
      for (int r = 0; r < 8; ++r) {
        int rit = wm * 32 + i * 16 + r + 8 * sel;
        float v = c[i][j][r] + bvl;
        if constexpr (MODE == 0) {
          outf[(size_t)(mbase + rit) * N + col] = v;
        } else if constexpr (MODE == 1 || MODE == 2) {
          int gr = mbase + rit;
          int b = gr >> 10, t = gr & 1023;       // T = 1024
          int h = col >> 6, hd = col & 63;       // HD = 64
          size_t o = (MODE == 1)
                         ? ((size_t)(b * 16 + h) * 1024 + t) * 64 + hd
                         : ((size_t)(b * 16 + h) * 64 + hd) * 1024 + t;
          outbf[o] = f2bf(v);
        } else if constexpr (MODE == 3) {
          if (rit < valid)
            outbf[(size_t)(mbase + rit) * N + col] = f2bf(fmaxf(v, 0.f));
        } else {  // MODE 4
          if (rit < valid)
            atomicAdd(outf + (size_t)rlLds[rit] * N + col, v * wlLds[rit]);
        }
      }
    }
}

// ---------------------------------------------------------------------------
// Flash attention with stoichiometric bias. One wave per 16-query tile.
// ---------------------------------------------------------------------------
__global__ __launch_bounds__(256) void attn_flash_kernel(
    const u16* __restrict__ Qh, const u16* __restrict__ Kh,
    const u16* __restrict__ Vt, const float* __restrict__ frac,
    const float* __restrict__ alphaP, u16* __restrict__ obf, int T) {
  __shared__ __attribute__((aligned(16))) unsigned char plds[8 * 16 * 80];
  const int tid = threadIdx.x, wid = tid >> 5, lane = tid & 31;
  const int sel = lane >> 4, ln = lane & 15;
  const int h = blockIdx.y, b = blockIdx.z;
  const int bh = b * 16 + h;
  const u16* Qb = Qh + (size_t)bh * T * 64;
  const u16* Kb = Kh + (size_t)bh * T * 64;
  const u16* Vb = Vt + (size_t)bh * 64 * T;
  const float* fb = frac + (size_t)b * T;
  const float alpha = alphaP[0];
  const int q0 = (blockIdx.x * 8 + wid) * 16;
  unsigned char* pb = plds + wid * (16 * 80);

  BfPack aq[2];
#pragma unroll
  for (int c2 = 0; c2 < 2; ++c2) {
    const u16* qp = Qb + (size_t)(q0 + ln) * 64 + c2 * 32 + sel * 8;
    aq[c2].q[0] = *(const i4*)qp;
    aq[c2].q[1] = *(const i4*)(qp + 16);
  }
  float fq[8];
#pragma unroll
  for (int r = 0; r < 8; ++r) fq[r] = fb[q0 + r + 8 * sel];

  v8f o[4] = {};
  float mrow[8], lrow[8];
#pragma unroll
  for (int r = 0; r < 8; ++r) { mrow[r] = -3.0e38f; lrow[r] = 0.f; }

  for (int kb = 0; kb < T; kb += 32) {
    v8f s0 = {}, s1 = {};
#pragma unroll
    for (int c2 = 0; c2 < 2; ++c2) {
      BfPack bk;
      const u16* kp = Kb + (size_t)(kb + ln) * 64 + c2 * 32 + sel * 8;
      bk.q[0] = *(const i4*)kp; bk.q[1] = *(const i4*)(kp + 16);
      s0 = wmma_bf16(aq[c2].v, bk.v, s0);
      const u16* kp1 = Kb + (size_t)(kb + 16 + ln) * 64 + c2 * 32 + sel * 8;
      bk.q[0] = *(const i4*)kp1; bk.q[1] = *(const i4*)(kp1 + 16);
      s1 = wmma_bf16(aq[c2].v, bk.v, s1);
    }
    const float fk0 = fb[kb + ln], fk1 = fb[kb + 16 + ln];
    float p0[8], p1[8];
#pragma unroll
    for (int r = 0; r < 8; ++r) {
      float d0 = fq[r] - fk0, d1 = fq[r] - fk1;
      float v0 = s0[r] * 0.125f +
                 alpha * __logf(1.f + fabsf(d0)) * (d0 >= 0.f ? 1.f : -1.f);
      float v1 = s1[r] * 0.125f +
                 alpha * __logf(1.f + fabsf(d1)) * (d1 >= 0.f ? 1.f : -1.f);
      float t = fmaxf(v0, v1);
      t = fmaxf(t, __shfl_xor(t, 1, 32));
      t = fmaxf(t, __shfl_xor(t, 2, 32));
      t = fmaxf(t, __shfl_xor(t, 4, 32));
      t = fmaxf(t, __shfl_xor(t, 8, 32));
      float mn   = fmaxf(mrow[r], t);
      float corr = __expf(mrow[r] - mn);
      float e0 = __expf(v0 - mn), e1 = __expf(v1 - mn);
      float rs = e0 + e1;
      rs += __shfl_xor(rs, 1, 32); rs += __shfl_xor(rs, 2, 32);
      rs += __shfl_xor(rs, 4, 32); rs += __shfl_xor(rs, 8, 32);
      lrow[r] = lrow[r] * corr + rs;
      mrow[r] = mn;
      p0[r] = e0; p1[r] = e1;
#pragma unroll
      for (int j = 0; j < 4; ++j) o[j][r] *= corr;
    }
#pragma unroll
    for (int r = 0; r < 8; ++r) {            // P: C-layout -> LDS (16x32 bf16)
      int rr = r + 8 * sel;
      *(u16*)(pb + rr * 80 + ln * 2)      = f2bf(p0[r]);
      *(u16*)(pb + rr * 80 + 32 + ln * 2) = f2bf(p1[r]);
    }
    asm volatile("s_wait_dscnt 0" ::: "memory");
    BfPack pa;
    {
      const unsigned char* pp = pb + ln * 80 + sel * 16;
      pa.q[0] = *(const i4*)pp;
      pa.q[1] = *(const i4*)(pp + 32);
    }
#pragma unroll
    for (int j = 0; j < 4; ++j) {
      BfPack bvv;
      const u16* vp = Vb + (size_t)(j * 16 + ln) * T + kb + sel * 8;
      bvv.q[0] = *(const i4*)vp; bvv.q[1] = *(const i4*)(vp + 16);
      o[j] = wmma_bf16(pa.v, bvv.v, o[j]);
    }
    asm volatile("s_wait_dscnt 0" ::: "memory");
  }
#pragma unroll
  for (int j = 0; j < 4; ++j)
#pragma unroll
    for (int r = 0; r < 8; ++r) {
      int t = q0 + r + 8 * sel;
      int d = h * 64 + j * 16 + ln;
      obf[((size_t)b * T + t) * 1024 + d] = f2bf(o[j][r] / lrow[r]);
    }
}

// ---------------------------------------------------------------------------
// out = LayerNorm(a + bsum) * g + beta   (D = 1024, one block per token)
// ---------------------------------------------------------------------------
__global__ __launch_bounds__(256) void add_ln_kernel(
    const float* __restrict__ a, const float* __restrict__ bsum,
    const float* __restrict__ g, const float* __restrict__ beta,
    float* __restrict__ outf, u16* __restrict__ outbf) {
  const int row = blockIdx.x, tid = threadIdx.x;
  const int wid = tid >> 5, lane = tid & 31;
  const float* ar = a + (size_t)row * 1024;
  const float* br = bsum ? bsum + (size_t)row * 1024 : nullptr;
  float v[4], s = 0.f;
#pragma unroll
  for (int i = 0; i < 4; ++i) {
    int c = tid + i * 256;
    v[i] = ar[c] + (br ? br[c] : 0.f);
    s += v[i];
  }
  __shared__ float red1[8], red2[8];
#pragma unroll
  for (int m = 1; m < 32; m <<= 1) s += __shfl_xor(s, m, 32);
  if (lane == 0) red1[wid] = s;
  __syncthreads();
  float tot = 0.f;
#pragma unroll
  for (int i = 0; i < 8; ++i) tot += red1[i];
  float mean = tot * (1.f / 1024.f);
  float s2 = 0.f;
#pragma unroll
  for (int i = 0; i < 4; ++i) { float d = v[i] - mean; s2 += d * d; }
#pragma unroll
  for (int m = 1; m < 32; m <<= 1) s2 += __shfl_xor(s2, m, 32);
  if (lane == 0) red2[wid] = s2;
  __syncthreads();
  float tot2 = 0.f;
#pragma unroll
  for (int i = 0; i < 8; ++i) tot2 += red2[i];
  float inv = rsqrtf(tot2 * (1.f / 1024.f) + 1e-5f);
#pragma unroll
  for (int i = 0; i < 4; ++i) {
    int c = tid + i * 256;
    float y = (v[i] - mean) * inv * g[c] + beta[c];
    outf[(size_t)row * 1024 + c] = y;
    if (outbf) outbf[(size_t)row * 1024 + c] = f2bf(y);
  }
}

// ---------------------------------------------------------------------------
// Gate: softmax(x @ gate_W + gate_b), top-2. One wave per token.
// ---------------------------------------------------------------------------
__global__ __launch_bounds__(256) void gate_topk_kernel(
    const float* __restrict__ x, const float* __restrict__ gW,
    const float* __restrict__ gb, float* __restrict__ s2, int* __restrict__ i2,
    int Ntok) {
  const int wid = threadIdx.x >> 5, lane = threadIdx.x & 31;
  const int n = blockIdx.x * 8 + wid;
  if (n >= Ntok) return;
  const float* xr = x + (size_t)n * 1024;
  float acc[8] = {0, 0, 0, 0, 0, 0, 0, 0};
  for (int i = lane; i < 1024; i += 32) {
    float xv = xr[i];
    const float* w = gW + (size_t)i * 8;
#pragma unroll
    for (int e = 0; e < 8; ++e) acc[e] += xv * w[e];
  }
#pragma unroll
  for (int e = 0; e < 8; ++e)
#pragma unroll
    for (int m = 1; m < 32; m <<= 1) acc[e] += __shfl_xor(acc[e], m, 32);
  if (lane == 0) {
    float p[8], mx = -3.0e38f;
#pragma unroll
    for (int e = 0; e < 8; ++e) { p[e] = acc[e] + gb[e]; mx = fmaxf(mx, p[e]); }
    float sum = 0.f;
#pragma unroll
    for (int e = 0; e < 8; ++e) { p[e] = __expf(p[e] - mx); sum += p[e]; }
    float is = 1.f / sum;
#pragma unroll
    for (int e = 0; e < 8; ++e) p[e] *= is;
    int e1 = 0;
    for (int e = 1; e < 8; ++e) if (p[e] > p[e1]) e1 = e;   // first max
    int e2 = (e1 == 0) ? 1 : 0;
    for (int e = 0; e < 8; ++e) if (e != e1 && p[e] > p[e2]) e2 = e;
    s2[2 * n] = p[e1]; s2[2 * n + 1] = p[e2];
    i2[2 * n] = e1;    i2[2 * n + 1] = e2;
  }
}

// ---------------------------------------------------------------------------
// MoE routing bookkeeping
// ---------------------------------------------------------------------------
__global__ __launch_bounds__(256) void count_kernel(const int* __restrict__ i2,
                                                    int* __restrict__ counts, int Ntok) {
  int n = blockIdx.x * 256 + threadIdx.x;
  if (n < Ntok) {
    atomicAdd(&counts[i2[2 * n]], 1);
    atomicAdd(&counts[i2[2 * n + 1]], 1);
  }
}

__global__ void scan_kernel(const int* __restrict__ counts, int* __restrict__ offs,
                            int* __restrict__ cursor) {
  if (threadIdx.x == 0) {
    int s = 0;
    for (int e = 0; e < 8; ++e) { offs[e] = s; cursor[e] = s; s += counts[e]; }
    offs[8] = s;
  }
}

__global__ __launch_bounds__(256) void scatter_kernel(
    const int* __restrict__ i2, const float* __restrict__ s2,
    int* __restrict__ cursor, int* __restrict__ rowlist,
    float* __restrict__ wlist, int Ntok) {
  int n = blockIdx.x * 256 + threadIdx.x;
  if (n >= Ntok) return;
  for (int s = 0; s < 2; ++s) {
    int e = i2[2 * n + s];
    int p = atomicAdd(&cursor[e], 1);
    rowlist[p] = n;
    wlist[p] = s2[2 * n + s];
  }
}

// ---------------------------------------------------------------------------
// host orchestration  (workspace requirement ~= 220 MB)
// ---------------------------------------------------------------------------
extern "C" void kernel_launch(void* const* d_in, const int* in_sizes, int n_in,
                              void* d_out, int out_size, void* d_ws, size_t ws_size,
                              hipStream_t stream) {
  const float* src    = (const float*)d_in[0];
  const float* frac   = (const float*)d_in[1];
  const float* Wq     = (const float*)d_in[2];
  const float* bq     = (const float*)d_in[3];
  const float* Wk     = (const float*)d_in[4];
  const float* bk     = (const float*)d_in[5];
  const float* Wv     = (const float*)d_in[6];
  const float* bv     = (const float*)d_in[7];
  const float* Wo     = (const float*)d_in[8];
  const float* bo     = (const float*)d_in[9];
  const float* alphaP = (const float*)d_in[10];
  const float* gW     = (const float*)d_in[11];
  const float* gb     = (const float*)d_in[12];
  const float* W1     = (const float*)d_in[13];
  const float* b1     = (const float*)d_in[14];
  const float* W2     = (const float*)d_in[15];
  const float* b2     = (const float*)d_in[16];
  const float* ln1g   = (const float*)d_in[17];
  const float* ln1b   = (const float*)d_in[18];
  const float* ln2g   = (const float*)d_in[19];
  const float* ln2b   = (const float*)d_in[20];

  const int D = 1024, T = 1024, F = 2048, E = 8;
  const int Bb = in_sizes[0] / (T * D);
  const int Ntok = Bb * T;

  char* base = (char*)d_ws;
  size_t off = 0;
  auto carve = [&](size_t bytes) -> void* {
    void* r = base + off;
    off = (off + bytes + 255) & ~(size_t)255;
    return r;
  };
  u16*   srcbf  = (u16*)carve((size_t)Ntok * D * 2);
  u16*   Wqt    = (u16*)carve((size_t)D * D * 2);
  u16*   Wkt    = (u16*)carve((size_t)D * D * 2);
  u16*   Wvt    = (u16*)carve((size_t)D * D * 2);
  u16*   Wot    = (u16*)carve((size_t)D * D * 2);
  u16*   W1t    = (u16*)carve((size_t)E * D * F * 2);
  u16*   W2t    = (u16*)carve((size_t)E * F * D * 2);
  u16*   Qh     = (u16*)carve((size_t)Ntok * D * 2);
  u16*   Kh     = (u16*)carve((size_t)Ntok * D * 2);
  u16*   Vt     = (u16*)carve((size_t)Ntok * D * 2);
  u16*   obf    = (u16*)carve((size_t)Ntok * D * 2);
  float* attno  = (float*)carve((size_t)Ntok * D * 4);
  float* x      = (float*)carve((size_t)Ntok * D * 4);
  u16*   xbf    = (u16*)carve((size_t)Ntok * D * 2);
  float* s2buf  = (float*)carve((size_t)Ntok * 2 * 4);
  int*   i2buf  = (int*)carve((size_t)Ntok * 2 * 4);
  int*   meta   = (int*)carve(64 * 4);     // counts[8] | offs[9] | cursor[8]
  int *counts = meta, *offs = meta + 8, *cursor = meta + 24;
  int*   rowlist= (int*)carve((size_t)2 * Ntok * 4);
  float* wlist  = (float*)carve((size_t)2 * Ntok * 4);
  u16*   hidbf  = (u16*)carve(((size_t)2 * Ntok + 128) * F * 2);
  float* moeout = (float*)carve((size_t)Ntok * D * 4);
  (void)ws_size; (void)n_in; (void)out_size;

  const dim3 blk(256);
  auto g1 = [](size_t n) { return dim3((unsigned)((n + 255) / 256)); };

  cvt_bf16_kernel<<<g1((size_t)Ntok * D), blk, 0, stream>>>(src, srcbf, (size_t)Ntok * D);
  transpose_bf16_kernel<<<g1((size_t)D * D), blk, 0, stream>>>(Wq, Wqt, D, D);
  transpose_bf16_kernel<<<g1((size_t)D * D), blk, 0, stream>>>(Wk, Wkt, D, D);
  transpose_bf16_kernel<<<g1((size_t)D * D), blk, 0, stream>>>(Wv, Wvt, D, D);
  transpose_bf16_kernel<<<g1((size_t)D * D), blk, 0, stream>>>(Wo, Wot, D, D);
  {
    dim3 gr((unsigned)(((size_t)D * F + 255) / 256), 1, E);
    transpose_bf16_kernel<<<gr, blk, 0, stream>>>(W1, W1t, D, F);
    transpose_bf16_kernel<<<gr, blk, 0, stream>>>(W2, W2t, F, D);
  }

  dim3 gproj(D / 128, Ntok / 128, 1);
  gemm_bf16_kernel<1, false><<<gproj, blk, 0, stream>>>(srcbf, Wqt, bq, nullptr, Qh,
      D, D, nullptr, nullptr, nullptr);
  gemm_bf16_kernel<1, false><<<gproj, blk, 0, stream>>>(srcbf, Wkt, bk, nullptr, Kh,
      D, D, nullptr, nullptr, nullptr);
  gemm_bf16_kernel<2, false><<<gproj, blk, 0, stream>>>(srcbf, Wvt, bv, nullptr, Vt,
      D, D, nullptr, nullptr, nullptr);

  attn_flash_kernel<<<dim3(T / 128, 16, Bb), blk, 0, stream>>>(Qh, Kh, Vt, frac,
      alphaP, obf, T);

  gemm_bf16_kernel<0, false><<<gproj, blk, 0, stream>>>(obf, Wot, bo, attno, nullptr,
      D, D, nullptr, nullptr, nullptr);

  add_ln_kernel<<<Ntok, blk, 0, stream>>>(src, attno, ln1g, ln1b, x, xbf);

  gate_topk_kernel<<<(Ntok + 7) / 8, blk, 0, stream>>>(x, gW, gb, s2buf, i2buf, Ntok);
  zero_f32_kernel<<<g1((size_t)Ntok * D), blk, 0, stream>>>(moeout, (size_t)Ntok * D);
  zero_f32_kernel<<<1, blk, 0, stream>>>((float*)meta, 64);
  count_kernel<<<g1((size_t)Ntok), blk, 0, stream>>>(i2buf, counts, Ntok);
  scan_kernel<<<1, 32, 0, stream>>>(counts, offs, cursor);
  scatter_kernel<<<g1((size_t)Ntok), blk, 0, stream>>>(i2buf, s2buf, cursor, rowlist,
                                                       wlist, Ntok);

  gemm_bf16_kernel<3, true><<<dim3(F / 128, (2 * Ntok) / 128, E), blk, 0, stream>>>(
      xbf, W1t, b1, nullptr, hidbf, F, D, offs, rowlist, nullptr);
  gemm_bf16_kernel<4, false><<<dim3(D / 128, (2 * Ntok) / 128, E), blk, 0, stream>>>(
      hidbf, W2t, b2, moeout, nullptr, D, F, offs, rowlist, wlist);

  add_ln_kernel<<<Ntok, blk, 0, stream>>>(x, moeout, ln2g, ln2b, (float*)d_out,
                                          nullptr);
}
